// MakeSpectrogram_71347996721533
// MI455X (gfx1250) — matrix operand order repaired and checked by reading, not compiled, not run on hardware
//
#include <hip/hip_runtime.h>

// ---------- CDNA5 WMMA vector types ----------
typedef __attribute__((ext_vector_type(16))) __bf16 v16bf;
typedef __attribute__((ext_vector_type(8)))  __bf16 v8bf;
typedef __attribute__((ext_vector_type(8)))  float  v8f;

#define FFT_SIZE   1024
#define HOP        256
#define BATCH      16
#define T_LEN      524288
#define N_FRAMES   2045              // (T_LEN - FFT_SIZE)/HOP + 1
#define N_FREQ     513               // FFT_SIZE/2 + 1
#define FREQ_TILES 33                // ceil(513/16)  (rows 513..527 of W exist, reads in range)
#define FRAMES_PER_WAVE  32          // two 16-frame WMMA tiles share one A fragment set
#define FRAMES_PER_BLOCK 128         // 4 waves x 32 frames
#define FRAME_BLOCKS 16              // ceil(2045/128)

// ---------- prepass: split f32 -> bf16 hi + bf16 lo (bf16x3 emulation) ----------
__global__ void split_f32_to_bf16x2(const float* __restrict__ src,
                                    __bf16* __restrict__ hi,
                                    __bf16* __restrict__ lo,
                                    int n) {
    int i = blockIdx.x * blockDim.x + threadIdx.x;
    if (i < n) {
        float f = src[i];
        __bf16 h = (__bf16)f;              // RNE
        hi[i] = h;
        lo[i] = (__bf16)(f - (float)h);    // residual
    }
}

// Load one 16x32 bf16 A-fragment (contiguous k order used by this file, same as B):
//   lane<16 : M=lane,    elems 0..15 -> K=0..15
//   lane>=16: M=lane-16, elems 0..15 -> K=16..31
__device__ __forceinline__ v16bf load_a_frag(const __bf16* rowPtr, int lane) {
    const int off0 = (lane < 16) ? 0 : 16;
    v8bf c0 = *(const v8bf*)(rowPtr + off0);
    v8bf c1 = *(const v8bf*)(rowPtr + off0 + 8);
    return __builtin_shufflevector(c0, c1, 0, 1, 2, 3, 4, 5, 6, 7,
                                           8, 9, 10, 11, 12, 13, 14, 15);
}

__device__ __forceinline__ v8f wmma_bf16(v16bf a, v16bf b, v8f c) {
    return __builtin_amdgcn_wmma_f32_16x16x32_bf16(
        /*neg_a=*/false, a, /*neg_b=*/false, b,
        /*c_mod=*/(short)0, c, /*reuse_a=*/false, /*reuse_b=*/false);
}

// ---------- main kernel: one wave = 16(freq) x 32(frames) = two WMMA tiles ----------
__global__ __launch_bounds__(128)
void spectro_wmma_bf16x3(const __bf16* __restrict__ xh, const __bf16* __restrict__ xl,
                         const __bf16* __restrict__ wrh, const __bf16* __restrict__ wrl,
                         const __bf16* __restrict__ wih, const __bf16* __restrict__ wil,
                         float* __restrict__ out) {
    const int lane  = threadIdx.x & 31;
    const int wave  = threadIdx.x >> 5;
    const int ktile = blockIdx.x;          // frequency tile
    const int fblk  = blockIdx.y;          // frame block (128 frames)
    const int batch = blockIdx.z;

    const int laneN    = lane & 15;
    const int freqBase = ktile * 16;

    // Two frame tiles per wave; clamp for ADDRESSING only (loads always in
    // bounds, no in-loop predication), guard the stores instead.
    const int frame0  = fblk * FRAMES_PER_BLOCK + wave * FRAMES_PER_WAVE + laneN;
    const int frame1  = frame0 + 16;
    const int frame0L = (frame0 < N_FRAMES) ? frame0 : (N_FRAMES - 1);
    const int frame1L = (frame1 < N_FRAMES) ? frame1 : (N_FRAMES - 1);

    // A rows: W[freqBase + (lane&15), :]  (freqBase+15 <= 527 < 1024)
    const int aRow = freqBase + laneN;
    const __bf16* prh = wrh + (size_t)aRow * FFT_SIZE;
    const __bf16* prl = wrl + (size_t)aRow * FFT_SIZE;
    const __bf16* pih = wih + (size_t)aRow * FFT_SIZE;
    const __bf16* pil = wil + (size_t)aRow * FFT_SIZE;

    // B: x[b, frame*HOP + k0 + K], K contiguous in memory.
    //   lane<16 : N=lane, K=0..15    lane>=16: N=lane-16, K=16..31
    const size_t xBase = (size_t)batch * T_LEN + ((lane < 16) ? 0 : 16);
    const __bf16* px0h = xh + xBase + (size_t)frame0L * HOP;
    const __bf16* px0l = xl + xBase + (size_t)frame0L * HOP;
    const __bf16* px1h = xh + xBase + (size_t)frame1L * HOP;
    const __bf16* px1l = xl + xBase + (size_t)frame1L * HOP;

    v8f accR0 = {}, accI0 = {};
    v8f accR1 = {}, accI1 = {};

    #pragma unroll
    for (int k0 = 0; k0 < FFT_SIZE; k0 += 32) {
        // shared A fragments (amortized over both frame tiles)
        v16bf arh = load_a_frag(prh + k0, lane);
        v16bf arl = load_a_frag(prl + k0, lane);
        v16bf aih = load_a_frag(pih + k0, lane);
        v16bf ail = load_a_frag(pil + k0, lane);

        v16bf b0h = *(const v16bf*)(px0h + k0);
        v16bf b0l = *(const v16bf*)(px0l + k0);
        v16bf b1h = *(const v16bf*)(px1h + k0);
        v16bf b1l = *(const v16bf*)(px1l + k0);

        // bf16x3: hi*hi + hi*lo + lo*hi, f32 accumulate
        accR0 = wmma_bf16(arh, b0h, accR0);
        accR0 = wmma_bf16(arh, b0l, accR0);
        accR0 = wmma_bf16(arl, b0h, accR0);
        accI0 = wmma_bf16(aih, b0h, accI0);
        accI0 = wmma_bf16(aih, b0l, accI0);
        accI0 = wmma_bf16(ail, b0h, accI0);

        accR1 = wmma_bf16(arh, b1h, accR1);
        accR1 = wmma_bf16(arh, b1l, accR1);
        accR1 = wmma_bf16(arl, b1h, accR1);
        accI1 = wmma_bf16(aih, b1h, accI1);
        accI1 = wmma_bf16(aih, b1l, accI1);
        accI1 = wmma_bf16(ail, b1h, accI1);
    }

    // D layout: VGPR r, lane<16 -> M=r, N=lane; lane>=16 -> M=r+8, N=lane-16
    const int mBase = (lane < 16) ? 0 : 8;
    const size_t outB = (size_t)batch * N_FREQ * N_FRAMES;
    #pragma unroll
    for (int r = 0; r < 8; ++r) {
        const int freq = freqBase + mBase + r;
        if (freq < N_FREQ) {
            const size_t rowOff = outB + (size_t)freq * N_FRAMES;
            if (frame0 < N_FRAMES) {
                const float re = accR0[r], im = accI0[r];
                out[rowOff + frame0] = re * re + im * im;
            }
            if (frame1 < N_FRAMES) {
                const float re = accR1[r], im = accI1[r];
                out[rowOff + frame1] = re * re + im * im;
            }
        }
    }
}

extern "C" void kernel_launch(void* const* d_in, const int* in_sizes, int n_in,
                              void* d_out, int out_size, void* d_ws, size_t ws_size,
                              hipStream_t stream) {
    const float* x  = (const float*)d_in[0];   // (16, 524288)
    const float* Wr = (const float*)d_in[1];   // (1024, 1024)
    const float* Wi = (const float*)d_in[2];   // (1024, 1024)
    float* out = (float*)d_out;                // (16, 513, 2045)

    const size_t W_ELEMS = (size_t)FFT_SIZE * FFT_SIZE;   // 1048576
    const size_t X_ELEMS = (size_t)BATCH * T_LEN;         // 8388608

    // workspace: bf16 hi/lo pairs (4*2MB for W, 2*16MB for x = 40MB)
    __bf16* wrh = (__bf16*)d_ws;
    __bf16* wrl = wrh + W_ELEMS;
    __bf16* wih = wrl + W_ELEMS;
    __bf16* wil = wih + W_ELEMS;
    __bf16* xh  = wil + W_ELEMS;
    __bf16* xl  = xh + X_ELEMS;

    split_f32_to_bf16x2<<<(unsigned)((W_ELEMS + 255) / 256), 256, 0, stream>>>(Wr, wrh, wrl, (int)W_ELEMS);
    split_f32_to_bf16x2<<<(unsigned)((W_ELEMS + 255) / 256), 256, 0, stream>>>(Wi, wih, wil, (int)W_ELEMS);
    split_f32_to_bf16x2<<<(unsigned)((X_ELEMS + 255) / 256), 256, 0, stream>>>(x, xh, xl, (int)X_ELEMS);

    dim3 grid(FREQ_TILES, FRAME_BLOCKS, BATCH);
    spectro_wmma_bf16x3<<<grid, 128, 0, stream>>>(xh, xl, wrh, wrl, wih, wil, out);
}
